// PointNetKAN_20358144983126
// MI455X (gfx1250) — matrix operand order repaired and checked by reading, not compiled, not run on hardware
//
#include <hip/hip_runtime.h>
#include <hip/hip_bf16.h>
#include <math.h>

// ---------------------------------------------------------------------------
// PointNet-KAN forward for MI455X (gfx1250, wave32, WMMA).
// GEMM precision: f16 A/B, f32 accumulate via v_wmma_f32_16x16x32_f16.
// BatchNorm folded into the next layer's on-the-fly poly featurization.
// B tiles streamed to LDS with GLOBAL_LOAD_ASYNC_TO_LDS_B128 when available.
// ---------------------------------------------------------------------------

typedef __attribute__((ext_vector_type(16))) _Float16 v16h;
typedef __attribute__((ext_vector_type(8)))  _Float16 v8h;
typedef __attribute__((ext_vector_type(4)))  _Float16 v4h;
typedef __attribute__((ext_vector_type(8)))  float    v8f;
typedef __attribute__((ext_vector_type(4)))  float    v4f;
typedef int v4i __attribute__((vector_size(16)));   // matches async-LDS builtin param

#define TM   128
#define TN   128
#define TK   32
#define LDT  40      // padded LDS row stride in halves (80B, keeps b128 chunks aligned)
#define NPTS 1024
#define MROWS (32 * NPTS)   // 32768 points total

#if __has_builtin(__builtin_amdgcn_global_load_async_to_lds_b128)
#define HAVE_ASYNC_LDS 1
#endif

__device__ __forceinline__ float fast_tanh(float x)
{
    // branch-free: tanh(x) = (e^{2x}-1)/(e^{2x}+1); clamp avoids inf*0
    x = fminf(9.f, fmaxf(-9.f, x));
    const float e = __expf(2.f * x);                 // v_exp_f32
    return (e - 1.f) * __builtin_amdgcn_rcpf(e + 1.f);
}

// ---------------------------------------------------------------------------
// Fused poly-featurize + WMMA GEMM.
//   A[M, K=4*(C0+C1)] built on the fly from raw activations + BN stats:
//     u = (act - mean[c]) * rstd[c];  t = tanh(u)
//     P0=1, P1=2t, P2=1.875*t*P1-0.75, P3=(28/15)*t*P2-0.8*P1   (Jacobi a=b=1)
//   B = Wt (Cout x K row-major, pre-transposed f16 weights)
//   act1 (CONCAT only) is a per-batch broadcast vector (B x C1) -> concat ch.
// Block: 256 thr (8 waves, 4x2), tile 128x128, wave tile 32x64 (8 WMMA/K-step)
// ---------------------------------------------------------------------------
template <bool CONCAT>
__global__ __launch_bounds__(256)
void kan_gemm(const _Float16* __restrict__ act0, const float* __restrict__ st0, int C0,
              const _Float16* __restrict__ act1, const float* __restrict__ st1, int C1,
              const _Float16* __restrict__ Wt,
              _Float16* __restrict__ outH, float* __restrict__ outF, int Cout)
{
    const int Ctot = CONCAT ? (C0 + C1) : C0;
    const int K    = 4 * Ctot;

    __shared__ alignas(16) _Float16 As[TM * LDT];
    __shared__ alignas(16) _Float16 Bs[TN * LDT];

    const int tid  = threadIdx.x;
    const int lane = tid & 31;
    const int wave = tid >> 5;
    const int wm   = wave & 3;           // wave M offset: wm*32
    const int wn   = wave >> 2;          // wave N offset: wn*64
    const int m0   = blockIdx.x * TM;
    const int n0   = blockIdx.y * TN;

    v8f acc[2][4] = {};

    // A-producer mapping: 256 thr -> 128 rows x 2 channel-quads
    const int am = tid >> 1;             // tile row 0..127
    const int ac = (tid & 1) * 4;        // channel quad within 8-channel K-slab
    const int gm = m0 + am;              // global point row
    const int bb = gm >> 10;             // batch index (1024 pts / cloud)

    // B-producer mapping: 256 thr -> 128 rows x 2 16-half chunks of K
    const int bn = tid >> 1;             // 0..127
    const int bk = (tid & 1) * 16;       // 0 or 16

    const int KT = (K + TK - 1) / TK;
    for (int kt = 0; kt < KT; ++kt) {
        const int kk0   = kt * TK;
        const int cbase = kk0 >> 2;      // first input channel of this K-slab

        // ---- produce A tile: normalized tanh -> Jacobi polys, f16 into LDS ----
        {
            const int c0f = cbase + ac;  // first channel of this quad
            v4h rv  = {};
            v4f muv = {}, rsv = {};
            bool vec = false;
            if (c0f + 4 <= C0) {
                rv  = *(const v4h*)&act0[(size_t)gm * C0 + c0f];
                muv = *(const v4f*)&st0[c0f];
                rsv = *(const v4f*)&st0[C0 + c0f];
                vec = true;
            } else if (CONCAT && c0f >= C0) {
                const int c1f = c0f - C0;
                if (c1f + 4 <= C1) {
                    rv  = *(const v4h*)&act1[(size_t)bb * C1 + c1f];
                    muv = *(const v4f*)&st1[c1f];
                    rsv = *(const v4f*)&st1[C1 + c1f];
                    vec = true;
                }
            }
            v8h lo = {}, hi = {};
            #pragma unroll
            for (int j = 0; j < 4; ++j) {
                const int c = c0f + j;
                float a = 0.f, mu = 0.f, rs = 0.f;
                bool valid = vec;
                if (vec) {
                    a = (float)rv[j]; mu = muv[j]; rs = rsv[j];
                } else if (c < Ctot) {
                    if (c < C0) {
                        a = (float)act0[(size_t)gm * C0 + c];
                        mu = st0[c]; rs = st0[C0 + c];
                    } else if (CONCAT) {
                        const int c1 = c - C0;
                        a = (float)act1[(size_t)bb * C1 + c1];
                        mu = st1[c1]; rs = st1[C1 + c1];
                    }
                    valid = true;
                }
                float p0 = 0.f, p1 = 0.f, p2 = 0.f, p3 = 0.f;
                if (valid) {
                    const float t = fast_tanh((a - mu) * rs);
                    p0 = 1.f;
                    p1 = 2.f * t;
                    p2 = 1.875f * t * p1 - 0.75f * p0;
                    p3 = (28.f / 15.f) * t * p2 - 0.8f * p1;
                }
                if (j < 2) {
                    lo[j * 4 + 0] = (_Float16)p0; lo[j * 4 + 1] = (_Float16)p1;
                    lo[j * 4 + 2] = (_Float16)p2; lo[j * 4 + 3] = (_Float16)p3;
                } else {
                    hi[(j - 2) * 4 + 0] = (_Float16)p0; hi[(j - 2) * 4 + 1] = (_Float16)p1;
                    hi[(j - 2) * 4 + 2] = (_Float16)p2; hi[(j - 2) * 4 + 3] = (_Float16)p3;
                }
            }
            v8h* dst = (v8h*)&As[am * LDT + ac * 4];   // 16B-aligned (80B stride)
            dst[0] = lo;
            dst[1] = hi;
        }

        // ---- produce B tile: Bs[n][k] = Wt[n0+n][kk0+k]  (row-major Cout x K) ----
        {
            const int row  = n0 + bn;
            const bool full = (row < Cout) && (kk0 + bk + 16 <= K);
            if (full) {
                const _Float16* gsrc = &Wt[(size_t)row * K + kk0 + bk];
                _Float16*       ldst = &Bs[bn * LDT + bk];
#ifdef HAVE_ASYNC_LDS
                __builtin_amdgcn_global_load_async_to_lds_b128(
                    (v4i*)gsrc,
                    (__attribute__((address_space(3))) v4i*)ldst, 0, 0);
                __builtin_amdgcn_global_load_async_to_lds_b128(
                    (v4i*)(gsrc + 8),
                    (__attribute__((address_space(3))) v4i*)(ldst + 8), 0, 0);
#else
                ((v8h*)ldst)[0] = *(const v8h*)gsrc;
                ((v8h*)ldst)[1] = *(const v8h*)(gsrc + 8);
#endif
                __builtin_prefetch(gsrc + TK, 0, 1);   // global_prefetch_b8
            } else {
                v8h w0 = {}, w1 = {};
                if (row < Cout) {
                    for (int j = 0; j < 16; ++j) {
                        const int k = kk0 + bk + j;
                        if (k < K) {
                            const _Float16 v = Wt[(size_t)row * K + k];
                            if (j < 8) w0[j] = v; else w1[j - 8] = v;
                        }
                    }
                }
                *(v8h*)&Bs[bn * LDT + bk]     = w0;
                *(v8h*)&Bs[bn * LDT + bk + 8] = w1;
            }
        }
#ifdef HAVE_ASYNC_LDS
#if __has_builtin(__builtin_amdgcn_s_wait_asynccnt)
        __builtin_amdgcn_s_wait_asynccnt(0);
#else
        asm volatile("s_wait_asynccnt 0x0" ::: "memory");
#endif
#endif
        __syncthreads();

        // ---- WMMA: each wave does 2x4 tiles of 16x16 over K=32 ----
        {
            const int kh = lane >> 4;    // lane half selects K sub-block
            const int ml = lane & 15;

            v16h afrag[2], bfrag[4];
            #pragma unroll
            for (int i = 0; i < 2; ++i) {
                const int row = wm * 32 + i * 16 + ml;
                // A layout: lanes<16 hold K {0..7,16..23}; lanes>=16 hold K {8..15,24..31}
                const v8h alo = *(const v8h*)&As[row * LDT + kh * 8];
                const v8h ahi = *(const v8h*)&As[row * LDT + kh * 8 + 16];
                #pragma unroll
                for (int j = 0; j < 8; ++j) { afrag[i][j] = alo[j]; afrag[i][8 + j] = ahi[j]; }
            }
            #pragma unroll
            for (int j2 = 0; j2 < 4; ++j2) {
                const int col = wn * 64 + j2 * 16 + ml;
                // B layout: lanes<16 hold K {0..7,16..23}; lanes>=16 hold K {8..15,24..31} (per column)
                const v8h blo = *(const v8h*)&Bs[col * LDT + kh * 8];
                const v8h bhi = *(const v8h*)&Bs[col * LDT + kh * 8 + 16];
                #pragma unroll
                for (int j = 0; j < 8; ++j) { bfrag[j2][j] = blo[j]; bfrag[j2][8 + j] = bhi[j]; }
            }
            #pragma unroll
            for (int i = 0; i < 2; ++i)
                #pragma unroll
                for (int j = 0; j < 4; ++j)
                    acc[i][j] = __builtin_amdgcn_wmma_f32_16x16x32_f16(
                        false, afrag[i], false, bfrag[j],
                        (short)0, acc[i][j], false, false);
        }
        __syncthreads();
    }

    // ---- epilogue: C/D layout -> row-major stores ----
    {
        const int ml = lane & 15;
        const int hh = lane >> 4;
        #pragma unroll
        for (int i = 0; i < 2; ++i) {
            #pragma unroll
            for (int j = 0; j < 4; ++j) {
                const int n = n0 + wn * 64 + j * 16 + ml;
                if (n < Cout) {
                    #pragma unroll
                    for (int r = 0; r < 8; ++r) {
                        const int m = m0 + wm * 32 + i * 16 + r + 8 * hh;
                        const float v = acc[i][j][r];
                        if (outH) outH[(size_t)m * Cout + n] = (_Float16)v;
                        if (outF) outF[(size_t)m * Cout + n] = v;
                    }
                }
            }
        }
    }
}

// ---------------------------------------------------------------------------
// Per-channel BN statistics over all M rows: st[c]=mean, st[C+c]=rsqrt(var+eps)
// ---------------------------------------------------------------------------
__global__ __launch_bounds__(256)
void kan_stats(const _Float16* __restrict__ act, float* __restrict__ st, int C, int M)
{
    __shared__ float ss[256], sq[256];
    const int c = blockIdx.x, t = threadIdx.x;
    float s = 0.f, s2 = 0.f;
    for (int r = t; r < M; r += 256) {
        const float v = (float)act[(size_t)r * C + c];
        s += v; s2 += v * v;
    }
    ss[t] = s; sq[t] = s2;
    __syncthreads();
    for (int k = 128; k > 0; k >>= 1) {
        if (t < k) { ss[t] += ss[t + k]; sq[t] += sq[t + k]; }
        __syncthreads();
    }
    if (t == 0) {
        const float mu  = ss[0] / (float)M;
        const float var = sq[0] / (float)M - mu * mu;
        st[c]     = mu;
        st[C + c] = rsqrtf(var + 1e-5f);
    }
}

// Max over points per (batch, channel). BN is monotone, so max commutes with it.
__global__ __launch_bounds__(256)
void kan_maxpool(const _Float16* __restrict__ act, _Float16* __restrict__ gmax, int C)
{
    const int b = blockIdx.x;
    const int c = blockIdx.y * 256 + threadIdx.x;
    if (c >= C) return;
    float m = -3.4e38f;
    for (int n = 0; n < NPTS; ++n)
        m = fmaxf(m, (float)act[((size_t)b * NPTS + n) * C + c]);
    gmax[(size_t)b * C + c] = (_Float16)m;
}

// w (Cin, Cout, 4) f32  ->  Wt (Cout, 4*Cin) f16 row-major
__global__ __launch_bounds__(256)
void kan_wcvt(const float* __restrict__ w, _Float16* __restrict__ wt, int Cin, int Cout)
{
    const int idx = blockIdx.x * 256 + threadIdx.x;
    const int total = Cin * Cout * 4;
    if (idx >= total) return;
    const int d = idx & 3;
    const int o = (idx >> 2) % Cout;
    const int i = (idx >> 2) / Cout;
    wt[(size_t)o * (4 * Cin) + i * 4 + d] = (_Float16)w[idx];
}

// x (B,2,NPTS) f32 channel-major -> h0 (M,2) f16 point-major
__global__ __launch_bounds__(256)
void kan_xcvt(const float* __restrict__ x, _Float16* __restrict__ h0)
{
    const int idx = blockIdx.x * 256 + threadIdx.x;
    if (idx >= MROWS * 2) return;
    const int m = idx >> 1, c = idx & 1;
    const int b = m >> 10, n = m & (NPTS - 1);
    h0[idx] = (_Float16)x[((size_t)b * 2 + c) * NPTS + n];
}

// Identity BN stats for the raw input layer
__global__ void kan_stid(float* st, int C)
{
    const int t = threadIdx.x;
    if (t < C) { st[t] = 0.f; st[C + t] = 1.f; }
}

// h10 (M,3) f32 point-major -> out (B,3,NPTS) f32 channel-major
__global__ __launch_bounds__(256)
void kan_finout(const float* __restrict__ h10, float* __restrict__ out)
{
    const int idx = blockIdx.x * 256 + threadIdx.x;
    if (idx >= 32 * 3 * NPTS) return;
    const int n = idx & (NPTS - 1);
    const int c = (idx >> 10) % 3;
    const int b = idx / (3 * NPTS);
    out[idx] = h10[((size_t)b * NPTS + n) * 3 + c];
}

// ---------------------------------------------------------------------------
extern "C" void kernel_launch(void* const* d_in, const int* in_sizes, int n_in,
                              void* d_out, int out_size, void* d_ws, size_t ws_size,
                              hipStream_t stream)
{
    (void)in_sizes; (void)n_in; (void)out_size; (void)ws_size;

    const float* x = (const float*)d_in[0];
    const float* w[10];
    for (int i = 0; i < 10; ++i) w[i] = (const float*)d_in[1 + i];

    static const int CIN [10] = {2, 64, 64, 64, 128, 1088, 512, 256, 128, 128};
    static const int COUT[10] = {64, 64, 64, 128, 1024, 512, 256, 128, 128, 3};

    char* ws = (char*)d_ws;
    size_t off = 0;
    auto alloc = [&](size_t bytes) -> size_t {
        size_t o = off; off = (off + bytes + 255) & ~(size_t)255; return o;
    };

    // Activation buffers (f16). bufA: h1,h3,h5,h7,h9   bufB: h4,h6,h8   h2 persists.
    _Float16* h0   = (_Float16*)(ws + alloc((size_t)MROWS * 2 * 2));
    _Float16* bufA = (_Float16*)(ws + alloc((size_t)MROWS * 1024 * 2));
    _Float16* bufB = (_Float16*)(ws + alloc((size_t)MROWS * 512 * 2));
    _Float16* h2p  = (_Float16*)(ws + alloc((size_t)MROWS * 64 * 2));
    _Float16* gmax = (_Float16*)(ws + alloc((size_t)32 * 1024 * 2));
    float*    h10f = (float*)   (ws + alloc((size_t)MROWS * 3 * 4));

    float* st[10];                               // st[0]=identity(2ch), st[l]=stats of h_l
    st[0] = (float*)(ws + alloc(2 * 2 * 4));
    for (int l = 1; l <= 9; ++l)
        st[l] = (float*)(ws + alloc((size_t)2 * COUT[l - 1] * 4));

    _Float16* wt[10];
    for (int l = 0; l < 10; ++l)
        wt[l] = (_Float16*)(ws + alloc((size_t)4 * CIN[l] * COUT[l] * 2));

    // ---- setup: input convert, identity stats, weight transpose+f16 ----
    kan_stid<<<1, 32, 0, stream>>>(st[0], 2);
    kan_xcvt<<<(MROWS * 2 + 255) / 256, 256, 0, stream>>>(x, h0);
    for (int l = 0; l < 10; ++l) {
        const int total = CIN[l] * COUT[l] * 4;
        kan_wcvt<<<(total + 255) / 256, 256, 0, stream>>>(w[l], wt[l], CIN[l], COUT[l]);
    }

    auto gemm = [&](const _Float16* a0, const float* s0, int C0,
                    const _Float16* W, _Float16* oh, float* of, int Co) {
        dim3 grid(MROWS / TM, (Co + TN - 1) / TN);
        kan_gemm<false><<<grid, 256, 0, stream>>>(a0, s0, C0, nullptr, nullptr, 0, W, oh, of, Co);
    };
    auto stats = [&](const _Float16* a, float* s, int C) {
        kan_stats<<<C, 256, 0, stream>>>(a, s, C, MROWS);
    };

    // ---- layer pipeline ----
    _Float16* h1 = bufA;
    gemm(h0,  st[0],   2, wt[0], h1,  nullptr, 64);    stats(h1,  st[1], 64);
    gemm(h1,  st[1],  64, wt[1], h2p, nullptr, 64);    stats(h2p, st[2], 64);
    _Float16* h3 = bufA;
    gemm(h2p, st[2],  64, wt[2], h3,  nullptr, 64);    stats(h3,  st[3], 64);
    _Float16* h4 = bufB;
    gemm(h3,  st[3],  64, wt[3], h4,  nullptr, 128);   stats(h4,  st[4], 128);
    _Float16* h5 = bufA;
    gemm(h4,  st[4], 128, wt[4], h5,  nullptr, 1024);  stats(h5,  st[5], 1024);
    kan_maxpool<<<dim3(32, 4), 256, 0, stream>>>(h5, gmax, 1024);
    _Float16* h6 = bufB;   // concat: local feature h2 (BN2) || broadcast BN5(max h5)
    {
        dim3 grid(MROWS / TM, (512 + TN - 1) / TN);
        kan_gemm<true><<<grid, 256, 0, stream>>>(h2p, st[2], 64, gmax, st[5], 1024,
                                                 wt[5], h6, nullptr, 512);
    }
    stats(h6, st[6], 512);
    _Float16* h7 = bufA;
    gemm(h6,  st[6], 512, wt[6], h7,  nullptr, 256);   stats(h7,  st[7], 256);
    _Float16* h8 = bufB;
    gemm(h7,  st[7], 256, wt[7], h8,  nullptr, 128);   stats(h8,  st[8], 128);
    _Float16* h9 = bufA;
    gemm(h8,  st[8], 128, wt[8], h9,  nullptr, 128);   stats(h9,  st[9], 128);
    gemm(h9,  st[9], 128, wt[9], nullptr, h10f, 3);    // no BN on final

    kan_finout<<<(32 * 3 * NPTS + 255) / 256, 256, 0, stream>>>(h10f, (float*)d_out);
}
